// RecurrentNeuralCollaborativeFiltering_22136261443928
// MI455X (gfx1250) — matrix operand order repaired and checked by reading, not compiled, not run on hardware
//
#include <hip/hip_runtime.h>

typedef __attribute__((ext_vector_type(16))) _Float16 v16h;
typedef __attribute__((ext_vector_type(8)))  _Float16 h8;
typedef __attribute__((ext_vector_type(8)))  float    v8f;

#define B_TOT   32768
#define U_CNT   50000
#define I_CNT   5000
#define E_DIM   32
#define H_DIM   64
#define M1_DIM  128
#define M2_DIM  64
#define L_STEPS 9
#define EPSV    1e-5f

// LDS layout (in halves):
//   [0,      12288)  w_ih as f16 (192 x 64, row-major)
//   [12288,  24576)  w_hh as f16 (192 x 64, row-major)
//   [24576,  32768)  per-wave h   (8 waves x 16 x 64)
//   [32768,  49152)  per-wave f1  (8 waves x 16 x 128)
#define WIH_OFF 0
#define WHH_OFF 12288
#define H_OFF   24576
#define F1_OFF  32768
#define LDS_HALVES 49152   // 96 KB

__device__ __forceinline__ v8f splat8(float x) {
  v8f r;
#pragma unroll
  for (int i = 0; i < 8; ++i) r[i] = x;
  return r;
}

__device__ __forceinline__ v8f wmma16(v16h a, v16h b, v8f c) {
  // D = A(16x32 f16) * B(32x16 f16) + C(16x16 f32)
  return __builtin_amdgcn_wmma_f32_16x16x32_f16(false, a, false, b, (short)0, c,
                                                false, false);
}

// Fast transcendentals: CDNA5 has native v_rcp_f32 / v_exp_f32 / v_tanh_f32.
__device__ __forceinline__ float sigmoid_fast(float x) {
  // 1/(1+exp(-x)) with hardware reciprocal (compiler tracks TRANS hazards).
  return __builtin_amdgcn_rcpf(1.0f + __expf(-x));
}

__device__ __forceinline__ float tanh_fast(float x) {
#if __has_builtin(__builtin_amdgcn_tanhf)
  return __builtin_amdgcn_tanhf(x);
#else
  float r;
  // v_nop covers the TRANS-op result hazard since the compiler can't see it.
  asm volatile("v_tanh_f32 %0, %1\n\tv_nop" : "=v"(r) : "v"(x));
  return r;
#endif
}

// A-fragment (16x32 f16) from a global f32 row of >=32 floats starting at
// feature `cbase` chunk. Contiguous K order: lane holds M=lane%16, halves 0..7 =
// K = cbase..cbase+7, halves 8..15 = K = cbase+8..cbase+15 (cbase already
// includes the (lane>>4)*16 shift).
__device__ __forceinline__ v16h afrag_g(const float* row, int cbase) {
  const float4* p = (const float4*)(row + cbase);
  const float4* q = (const float4*)(row + cbase + 8);
  float4 a = p[0], b = p[1], c = q[0], d = q[1];
  v16h r;
  r[0]  = (_Float16)a.x; r[1]  = (_Float16)a.y; r[2]  = (_Float16)a.z; r[3]  = (_Float16)a.w;
  r[4]  = (_Float16)b.x; r[5]  = (_Float16)b.y; r[6]  = (_Float16)b.z; r[7]  = (_Float16)b.w;
  r[8]  = (_Float16)c.x; r[9]  = (_Float16)c.y; r[10] = (_Float16)c.z; r[11] = (_Float16)c.w;
  r[12] = (_Float16)d.x; r[13] = (_Float16)d.y; r[14] = (_Float16)d.z; r[15] = (_Float16)d.w;
  return r;
}

// Same A-fragment but from an f16 row in LDS (two ds_load_b128).
__device__ __forceinline__ v16h afrag_l(const _Float16* row, int cbase) {
  h8 lo = *(const h8*)(row + cbase);
  h8 ho = *(const h8*)(row + cbase + 8);
  v16h r;
#pragma unroll
  for (int j = 0; j < 8; ++j) { r[j] = lo[j]; r[j + 8] = ho[j]; }
  return r;
}

// B-fragment (32x16 f16) for GEMM X @ W^T, W is (N x K) row-major f16 in LDS.
// Lane holds column n (row of W); 16 contiguous K halves at kpos
// (kpos already includes ktile*32 + (lane>>4)*16).
__device__ __forceinline__ v16h bfrag_l(const _Float16* W, int ld, int n, int kpos) {
  const h8* p = (const h8*)(W + n * ld + kpos);
  h8 lo = p[0], ho = p[1];
  v16h r;
#pragma unroll
  for (int j = 0; j < 8; ++j) { r[j] = lo[j]; r[j + 8] = ho[j]; }
  return r;
}

// B-fragment from a global f32 (N x K) row-major matrix, convert on the fly.
__device__ __forceinline__ v16h bfrag_gf(const float* W, int ld, int n, int kpos) {
  const float4* p = (const float4*)(W + n * ld + kpos);
  float4 a = p[0], b = p[1], c = p[2], d = p[3];
  v16h r;
  r[0]  = (_Float16)a.x; r[1]  = (_Float16)a.y; r[2]  = (_Float16)a.z; r[3]  = (_Float16)a.w;
  r[4]  = (_Float16)b.x; r[5]  = (_Float16)b.y; r[6]  = (_Float16)b.z; r[7]  = (_Float16)b.w;
  r[8]  = (_Float16)c.x; r[9]  = (_Float16)c.y; r[10] = (_Float16)c.z; r[11] = (_Float16)c.w;
  r[12] = (_Float16)d.x; r[13] = (_Float16)d.y; r[14] = (_Float16)d.z; r[15] = (_Float16)d.w;
  return r;
}

extern "C" __global__ __launch_bounds__(256)
void rncf_fused_kernel(const int* __restrict__ time_id, const int* __restrict__ user_id,
                       const int* __restrict__ item_id,
                       const float* __restrict__ user_emb, const float* __restrict__ item_emb,
                       const float* __restrict__ w_ih, const float* __restrict__ w_hh,
                       const float* __restrict__ b_ih, const float* __restrict__ b_hh,
                       const float* __restrict__ w1, const float* __restrict__ b1,
                       const float* __restrict__ g1, const float* __restrict__ be1,
                       const float* __restrict__ m1, const float* __restrict__ v1,
                       const float* __restrict__ w2, const float* __restrict__ b2,
                       const float* __restrict__ g2, const float* __restrict__ be2,
                       const float* __restrict__ m2, const float* __restrict__ v2,
                       const float* __restrict__ fc_w, float* __restrict__ out)
{
  extern __shared__ _Float16 lds[];
  _Float16* Wih = lds + WIH_OFF;
  _Float16* Whh = lds + WHH_OFF;

  const int tid  = threadIdx.x;
  const int wave = tid >> 5;
  const int lane = tid & 31;
  const int lh   = lane & 15;   // sample-in-tile (A/M) and n-in-tile (B/N,C)
  const int half = lane >> 4;   // half-wave select
  const int hi8  = half * 8;    // C-layout sample base for this lane
  const int hk0  = half * 16;   // B-frag K offset, ktile 0
  const int hk1  = 32 + half * 16;  // B-frag K offset, ktile 1
  const int c0a  = half * 16;   // A-frag chunk base

  _Float16* Hbuf  = lds + H_OFF  + wave * (16 * H_DIM);    // 16 x 64
  _Float16* F1buf = lds + F1_OFF + wave * (16 * M1_DIM);   // 16 x 128

  // ---- stage GRU weights as f16 into LDS (f32 -> f16), 4 elems/thread/iter
  for (int i = tid * 4; i < 192 * 64; i += 256 * 4) {
    float4 a = *(const float4*)(w_ih + i);
    float4 b = *(const float4*)(w_hh + i);
    Wih[i + 0] = (_Float16)a.x; Wih[i + 1] = (_Float16)a.y;
    Wih[i + 2] = (_Float16)a.z; Wih[i + 3] = (_Float16)a.w;
    Whh[i + 0] = (_Float16)b.x; Whh[i + 1] = (_Float16)b.y;
    Whh[i + 2] = (_Float16)b.z; Whh[i + 3] = (_Float16)b.w;
  }
  // ---- zero this wave's h buffer (h0 = 0)
  {
    h8 z = {};
    for (int i = lane; i < (16 * H_DIM) / 8; i += 32) ((h8*)Hbuf)[i] = z;
  }
  __syncthreads();

  // ---- per-lane sample (lanes l and l+16 share sample lh)
  const int S   = blockIdx.x * 128 + wave * 16 + lh;
  const int t0  = time_id[S];
  const int uid = user_id[S];
  const int iid = item_id[S];

  // ---- per-lane gate biases (feature n = tile*16 + lh)
  float brz[8], bin[4], bhn[4];
#pragma unroll
  for (int nt = 0; nt < 8; ++nt) {
    int n = nt * 16 + lh;
    brz[nt] = b_ih[n] + b_hh[n];
  }
#pragma unroll
  for (int j = 0; j < 4; ++j) {
    int n = 128 + j * 16 + lh;
    bin[j] = b_ih[n];
    bhn[j] = b_hh[n];
  }

  // ---- h in C layout: hc[j][i] = h[sample i+hi8][feature j*16+lh]
  v8f hc[4];
#pragma unroll
  for (int j = 0; j < 4; ++j) hc[j] = splat8(0.0f);

  // ================= GRU recurrence =================
  for (int t = 0; t < L_STEPS; ++t) {
    const float* urow = user_emb + ((size_t)(t0 + t) * U_CNT + (size_t)uid) * E_DIM;
    const float* irow = item_emb + ((size_t)(t0 + t) * I_CNT + (size_t)iid) * E_DIM;
    v16h ax0 = afrag_g(urow, c0a);            // X features 0..31  (user)
    v16h ax1 = afrag_g(irow, c0a);            // X features 32..63 (item)
    const _Float16* hrow = Hbuf + lh * H_DIM;
    v16h ah0 = afrag_l(hrow, c0a);            // h features 0..31
    v16h ah1 = afrag_l(hrow, 32 + c0a);       // h features 32..63

#pragma unroll
    for (int j = 0; j < 4; ++j) {
      const int nj = j * 16 + lh;
      // r-gate: gi_r + gh_r (+ both biases)
      v8f cr = splat8(brz[j]);
      cr = wmma16(ax0, bfrag_l(Wih, H_DIM, nj, hk0), cr);
      cr = wmma16(ax1, bfrag_l(Wih, H_DIM, nj, hk1), cr);
      cr = wmma16(ah0, bfrag_l(Whh, H_DIM, nj, hk0), cr);
      cr = wmma16(ah1, bfrag_l(Whh, H_DIM, nj, hk1), cr);
      // z-gate
      v8f cz = splat8(brz[4 + j]);
      cz = wmma16(ax0, bfrag_l(Wih, H_DIM, 64 + nj, hk0), cz);
      cz = wmma16(ax1, bfrag_l(Wih, H_DIM, 64 + nj, hk1), cz);
      cz = wmma16(ah0, bfrag_l(Whh, H_DIM, 64 + nj, hk0), cz);
      cz = wmma16(ah1, bfrag_l(Whh, H_DIM, 64 + nj, hk1), cz);
      // n-gate halves kept separate: gi_n, gh_n
      v8f cn = splat8(bin[j]);
      cn = wmma16(ax0, bfrag_l(Wih, H_DIM, 128 + nj, hk0), cn);
      cn = wmma16(ax1, bfrag_l(Wih, H_DIM, 128 + nj, hk1), cn);
      v8f ch = splat8(bhn[j]);
      ch = wmma16(ah0, bfrag_l(Whh, H_DIM, 128 + nj, hk0), ch);
      ch = wmma16(ah1, bfrag_l(Whh, H_DIM, 128 + nj, hk1), ch);

      // elementwise GRU update for this 16-feature slice
#pragma unroll
      for (int i = 0; i < 8; ++i) {
        float r  = sigmoid_fast(cr[i]);
        float z  = sigmoid_fast(cz[i]);
        float nn = tanh_fast(cn[i] + r * ch[i]);
        float h  = (1.0f - z) * nn + z * hc[j][i];
        hc[j][i] = h;
        // mirror to LDS (f16) for next step's A-fragment load
        Hbuf[(i + hi8) * H_DIM + nj] = (_Float16)h;
      }
    }
    // within-wave LDS is in-order; this just stops compiler reordering
    asm volatile("" ::: "memory");
  }

  // ================= MLP head =================
  // f1 = relu(bn1(h @ w1^T + b1))  -> 16 x 128
  {
    const _Float16* hrow = Hbuf + lh * H_DIM;
    v16h ah0 = afrag_l(hrow, c0a);
    v16h ah1 = afrag_l(hrow, 32 + c0a);
#pragma unroll
    for (int nt = 0; nt < 8; ++nt) {
      int n = nt * 16 + lh;
      v8f c = splat8(0.0f);
      c = wmma16(ah0, bfrag_gf(w1, H_DIM, n, hk0), c);
      c = wmma16(ah1, bfrag_gf(w1, H_DIM, n, hk1), c);
      float sc = g1[n] * rsqrtf(v1[n] + EPSV);
      float sh = (b1[n] - m1[n]) * sc + be1[n];
#pragma unroll
      for (int i = 0; i < 8; ++i) {
        float f = fmaxf(c[i] * sc + sh, 0.0f);
        F1buf[(i + hi8) * M1_DIM + n] = (_Float16)f;
      }
    }
    asm volatile("" ::: "memory");
  }

  // f2 = relu(bn2(f1 @ w2^T + b2)); y = f2 @ fc_w^T
  {
    const _Float16* frow = F1buf + lh * M1_DIM;
    v16h af0 = afrag_l(frow, 0 * 32 + c0a);
    v16h af1 = afrag_l(frow, 1 * 32 + c0a);
    v16h af2 = afrag_l(frow, 2 * 32 + c0a);
    v16h af3 = afrag_l(frow, 3 * 32 + c0a);

    float acc[8];
#pragma unroll
    for (int i = 0; i < 8; ++i) acc[i] = 0.0f;

#pragma unroll
    for (int nt = 0; nt < 4; ++nt) {
      int n = nt * 16 + lh;
      v8f c = splat8(0.0f);
      c = wmma16(af0, bfrag_gf(w2, M1_DIM, n, 0 * 32 + half * 16), c);
      c = wmma16(af1, bfrag_gf(w2, M1_DIM, n, 1 * 32 + half * 16), c);
      c = wmma16(af2, bfrag_gf(w2, M1_DIM, n, 2 * 32 + half * 16), c);
      c = wmma16(af3, bfrag_gf(w2, M1_DIM, n, 3 * 32 + half * 16), c);
      float sc = g2[n] * rsqrtf(v2[n] + EPSV);
      float sh = (b2[n] - m2[n]) * sc + be2[n];
      float fw = fc_w[n];
#pragma unroll
      for (int i = 0; i < 8; ++i) {
        float f = fmaxf(c[i] * sc + sh, 0.0f);
        acc[i] += f * fw;
      }
    }

    // reduce the 16 feature-lanes of each half-wave (samples stay separated)
#pragma unroll
    for (int m = 1; m < 16; m <<= 1) {
#pragma unroll
      for (int i = 0; i < 8; ++i) acc[i] += __shfl_xor(acc[i], m, 32);
    }
    if (lh == 0) {
#pragma unroll
      for (int i = 0; i < 8; ++i)
        out[blockIdx.x * 128 + wave * 16 + hi8 + i] = acc[i];
    }
  }
}

extern "C" void kernel_launch(void* const* d_in, const int* in_sizes, int n_in,
                              void* d_out, int out_size, void* d_ws, size_t ws_size,
                              hipStream_t stream) {
  (void)in_sizes; (void)n_in; (void)d_ws; (void)ws_size; (void)out_size;
  const int*   time_id  = (const int*)  d_in[0];
  const int*   user_id  = (const int*)  d_in[1];
  const int*   item_id  = (const int*)  d_in[2];
  const float* user_emb = (const float*)d_in[3];
  const float* item_emb = (const float*)d_in[4];
  const float* w_ih     = (const float*)d_in[5];
  const float* w_hh     = (const float*)d_in[6];
  const float* b_ih     = (const float*)d_in[7];
  const float* b_hh     = (const float*)d_in[8];
  const float* w1       = (const float*)d_in[9];
  const float* b1       = (const float*)d_in[10];
  const float* g1       = (const float*)d_in[11];
  const float* be1      = (const float*)d_in[12];
  const float* m1       = (const float*)d_in[13];
  const float* v1       = (const float*)d_in[14];
  const float* w2       = (const float*)d_in[15];
  const float* b2       = (const float*)d_in[16];
  const float* g2       = (const float*)d_in[17];
  const float* be2      = (const float*)d_in[18];
  const float* m2       = (const float*)d_in[19];
  const float* v2       = (const float*)d_in[20];
  const float* fc_w     = (const float*)d_in[21];
  float* out = (float*)d_out;

  const int blocks = B_TOT / 128;              // 256 blocks, 128 samples each
  const size_t shmem = LDS_HALVES * sizeof(_Float16);  // 96 KB (CDNA5: 320 KB/WGP)
  rncf_fused_kernel<<<blocks, 256, shmem, stream>>>(
      time_id, user_id, item_id, user_emb, item_emb,
      w_ih, w_hh, b_ih, b_hh,
      w1, b1, g1, be1, m1, v1,
      w2, b2, g2, be2, m2, v2,
      fc_w, out);
}
